// DeltaVisionMambaBlock_40939628266024
// MI455X (gfx1250) — matrix operand order repaired and checked by reading, not compiled, not run on hardware
//
#include <hip/hip_runtime.h>
#include <hip/hip_bf16.h>
#include <math.h>

// ---- problem constants (match reference) ----
#define CB   2
#define CN   2048
#define CD   768
#define CDI  1536
#define CDS  16
#define CDC  4
#define CDTR 48
#define CEPS 1e-5f
#define CM   (CB * CN)           // 4096 flattened tokens

typedef __attribute__((ext_vector_type(16))) __bf16       v16bf;
typedef __attribute__((ext_vector_type(8)))  float        v8f;
typedef __attribute__((ext_vector_type(4)))  unsigned int v4u;
typedef __attribute__((ext_vector_type(8)))  int          v8i;
typedef __attribute__((ext_vector_type(4)))  int          v4i;

// ---- bf16 fragment packing helpers ----
__device__ __forceinline__ v16bf pack16(float4 a, float4 b, float4 c, float4 d) {
    v16bf v;
    v[0]  = (__bf16)a.x; v[1]  = (__bf16)a.y; v[2]  = (__bf16)a.z; v[3]  = (__bf16)a.w;
    v[4]  = (__bf16)b.x; v[5]  = (__bf16)b.y; v[6]  = (__bf16)b.z; v[7]  = (__bf16)b.w;
    v[8]  = (__bf16)c.x; v[9]  = (__bf16)c.y; v[10] = (__bf16)c.z; v[11] = (__bf16)c.w;
    v[12] = (__bf16)d.x; v[13] = (__bf16)d.y; v[14] = (__bf16)d.z; v[15] = (__bf16)d.w;
    return v;
}
__device__ __forceinline__ float4 ld4(const float* p) { return *reinterpret_cast<const float4*>(p); }

__device__ __forceinline__ void load8_cvt(v16bf& v, int ofs, const float* __restrict__ p, bool valid) {
    if (valid) {
        const float4 f0 = ld4(p);
        const float4 f1 = ld4(p + 4);
        v[ofs + 0] = (__bf16)f0.x; v[ofs + 1] = (__bf16)f0.y;
        v[ofs + 2] = (__bf16)f0.z; v[ofs + 3] = (__bf16)f0.w;
        v[ofs + 4] = (__bf16)f1.x; v[ofs + 5] = (__bf16)f1.y;
        v[ofs + 6] = (__bf16)f1.z; v[ofs + 7] = (__bf16)f1.w;
    } else {
        #pragma unroll
        for (int i = 0; i < 8; ++i) v[ofs + i] = (__bf16)0.0f;
    }
}

// ---- Tensor Data Mover: 2D tile (tile_d1 rows x tile_d0 elems, f32) -> LDS ----
// D# per CDNA5 ISA: group0 {count, lds_addr, global_addr, type=2},
// group1 {data_size=4B, tensor dims, tile dims, dim0 stride}. Groups 2/3 (+extra group
// on the 6-arg clang-23 builtin) are NULL-equivalent zeros for a 2D tile.
__device__ __forceinline__ void tdm_load_2d_f32(unsigned lds_off, const float* gptr,
                                                unsigned tile_d0, unsigned tile_d1,
                                                unsigned tens_d0, unsigned tens_d1,
                                                unsigned stride_d0) {
    const unsigned long long ga = (unsigned long long)(size_t)gptr;
    v4u g0;
    g0.x = 1u;                                                    // count=1, user mode
    g0.y = lds_off;                                               // lds_addr [63:32]
    g0.z = (unsigned)(ga & 0xFFFFFFFFu);                          // global_addr [95:64]
    g0.w = (unsigned)((ga >> 32) & 0x01FFFFFFu) | 0x80000000u;    // addr [120:96] | type=2
    v8i g1;
    g1[0] = (int)(2u << 16);                                      // data_size=4B, wg_mask=0
    g1[1] = (int)((tens_d0 & 0xFFFFu) << 16);                     // tensor_dim0 lo16
    g1[2] = (int)((tens_d0 >> 16) | ((tens_d1 & 0xFFFFu) << 16)); // dim0 hi16 | dim1 lo16
    g1[3] = (int)((tens_d1 >> 16) | (tile_d0 << 16));             // dim1 hi16 | tile_dim0
    g1[4] = (int)(tile_d1 & 0xFFFFu);                             // tile_dim1 (tile_dim2=0)
    g1[5] = (int)stride_d0;                                       // tensor_dim0_stride lo32
    g1[6] = 0;                                                    // stride hi16 | dim1_stride lo16
    g1[7] = 0;
    const v4i z4 = {0, 0, 0, 0};
    const v8i z8 = {0, 0, 0, 0, 0, 0, 0, 0};
    __builtin_amdgcn_tensor_load_to_lds(g0, g1, z4, z4, z8, 0);
}

// ============================================================================
// LDS-tiled WMMA GEMM via TDM staging. Requires M%64==0, Nc%128==0, K%32==0.
// Block = 256 thr = 8 waves in 2x4; each wave: 2x2 WMMA tiles (32x32 output).
// ============================================================================
template<bool HAS_BIAS, bool HAS_RESID>
__global__ __launch_bounds__(256)
void gemm_tdm_wmma(const float* __restrict__ A, int lda,
                   const float* __restrict__ W, int ldw,
                   float* __restrict__ C, int ldc,
                   const float* __restrict__ bias,
                   const float* __restrict__ resid,
                   int M, int Nc, int K) {
    __shared__ float ldsA[64 * 32];    // 8 KB
    __shared__ float ldsB[128 * 32];   // 16 KB
    const int wave  = threadIdx.x >> 5;
    const int lane  = threadIdx.x & 31;
    const int half  = lane >> 4;
    const int l16   = lane & 15;
    const int nBN   = Nc / 128;
    const int m0    = (blockIdx.x / nBN) * 64;
    const int n0    = (blockIdx.x % nBN) * 128;
    const int waveM = wave >> 2;       // 0..1
    const int waveN = wave & 3;        // 0..3

    v8f acc[2][2];
    #pragma unroll
    for (int i = 0; i < 2; ++i)
        #pragma unroll
        for (int j = 0; j < 2; ++j)
            acc[i][j] = (v8f){0.f, 0.f, 0.f, 0.f, 0.f, 0.f, 0.f, 0.f};

    const unsigned ldsAoff = (unsigned)(size_t)(&ldsA[0]);
    const unsigned ldsBoff = (unsigned)(size_t)(&ldsB[0]);

    for (int k0 = 0; k0 < K; k0 += 32) {
        if (wave == 0) {   // one wave issues both DMAs, waits TENSORcnt, then all sync
            tdm_load_2d_f32(ldsAoff, A + (size_t)m0 * lda + k0, 32u, 64u,
                            (unsigned)(K - k0), (unsigned)(M - m0), (unsigned)lda);
            tdm_load_2d_f32(ldsBoff, W + (size_t)n0 * ldw + k0, 32u, 128u,
                            (unsigned)(K - k0), (unsigned)(Nc - n0), (unsigned)ldw);
            __builtin_amdgcn_s_wait_tensorcnt(0);
        }
        __syncthreads();

        v16bf af[2], bfr[2];
        #pragma unroll
        for (int i = 0; i < 2; ++i) {
            // A-frag 16x32: lanes0-15 K {0..7,16..23}, lanes16-31 K {8..15,24..31}
            const float* rp = &ldsA[(waveM * 32 + i * 16 + l16) * 32];
            const int kb = half ? 8 : 0;
            af[i] = pack16(ld4(rp + kb), ld4(rp + kb + 4),
                           ld4(rp + 16 + kb), ld4(rp + 16 + kb + 4));
        }
        #pragma unroll
        for (int j = 0; j < 2; ++j) {
            // B-frag 32x16: n = l16; lanes0-15 K {0..7,16..23}, lanes16-31 K {8..15,24..31}
            const float* rp = &ldsB[(waveN * 32 + j * 16 + l16) * 32];
            const int kb = half ? 8 : 0;
            bfr[j] = pack16(ld4(rp + kb), ld4(rp + kb + 4),
                            ld4(rp + 16 + kb), ld4(rp + 16 + kb + 4));
        }
        #pragma unroll
        for (int i = 0; i < 2; ++i)
            #pragma unroll
            for (int j = 0; j < 2; ++j)
                acc[i][j] = __builtin_amdgcn_wmma_f32_16x16x32_bf16(
                    false, af[i], false, bfr[j], (short)0, acc[i][j], false, false);
        __syncthreads();
    }

    #pragma unroll
    for (int i = 0; i < 2; ++i) {
        #pragma unroll
        for (int j = 0; j < 2; ++j) {
            const int   n    = n0 + waveN * 32 + j * 16 + l16;
            const float bval = HAS_BIAS ? bias[n] : 0.f;
            #pragma unroll
            for (int r = 0; r < 8; ++r) {
                const int m = m0 + waveM * 32 + i * 16 + r + (half ? 8 : 0);
                float v = acc[i][j][r] + bval;
                if (HAS_RESID) v += resid[(size_t)m * ldc + n];
                C[(size_t)m * ldc + n] = v;
            }
        }
    }
}

// ============================================================================
// Direct-from-global WMMA GEMM for ragged shapes (Nc=80, K=48). One 16x16 tile
// per wave; peeled tail so the main K-loop has no EXEC-masked guards.
// ============================================================================
template<bool HAS_BIAS, bool SOFTPLUS_ACT>
__global__ __launch_bounds__(256)
void gemm_wmma_direct(const float* __restrict__ A, int lda,
                      const float* __restrict__ W, int ldw,
                      float* __restrict__ C, int ldc,
                      const float* __restrict__ bias,
                      int M, int Nc, int K) {
    const int wave   = threadIdx.x >> 5;
    const int lane   = threadIdx.x & 31;
    const int tilesN = Nc >> 4;
    const int tile   = blockIdx.x * 8 + wave;
    if (tile >= (M >> 4) * tilesN) return;       // wave-uniform
    const int tM   = tile / tilesN;
    const int tN   = tile % tilesN;
    const int half = lane >> 4;
    const int l16  = lane & 15;

    const float* pa = A + (size_t)(tM * 16 + l16) * lda;
    const float* pw = W + (size_t)(tN * 16 + l16) * ldw;

    v8f acc = {0.f, 0.f, 0.f, 0.f, 0.f, 0.f, 0.f, 0.f};
    int k0 = 0;
    for (; k0 + 32 <= K; k0 += 32) {             // unguarded main loop
        const int kA = k0 + (half ? 8 : 0);
        const int kB = k0 + (half ? 8 : 0);
        const v16bf a = pack16(ld4(pa + kA), ld4(pa + kA + 4),
                               ld4(pa + kA + 16), ld4(pa + kA + 20));
        const v16bf b = pack16(ld4(pw + kB), ld4(pw + kB + 4),
                               ld4(pw + kB + 16), ld4(pw + kB + 20));
        acc = __builtin_amdgcn_wmma_f32_16x16x32_bf16(false, a, false, b,
                                                      (short)0, acc, false, false);
    }
    if (k0 < K) {                                 // single guarded tail (K%32==16)
        const int kA0 = k0 + (half ? 8 : 0);
        const int kA1 = k0 + 16 + (half ? 8 : 0);
        const int kB0 = k0 + (half ? 8 : 0);
        v16bf a, b;
        load8_cvt(a, 0, pa + kA0, kA0 < K);
        load8_cvt(a, 8, pa + kA1, kA1 < K);
        load8_cvt(b, 0, pw + kB0, kB0 < K);
        load8_cvt(b, 8, pw + kB0 + 16, (kB0 + 16) < K);
        acc = __builtin_amdgcn_wmma_f32_16x16x32_bf16(false, a, false, b,
                                                      (short)0, acc, false, false);
    }

    const int   n    = tN * 16 + l16;
    const float bval = HAS_BIAS ? bias[n] : 0.f;
    #pragma unroll
    for (int r = 0; r < 8; ++r) {
        const int m = tM * 16 + r + (half ? 8 : 0);
        float v = acc[r] + bval;
        if (SOFTPLUS_ACT) v = (v > 20.f) ? v : log1pf(__expf(v));
        C[(size_t)m * ldc + n] = v;
    }
}

// ---- elementwise / reduction kernels ----
__global__ void diff_concat_kernel(const float* __restrict__ x, float* __restrict__ comb) {
    const int idx = blockIdx.x * blockDim.x + threadIdx.x;
    if (idx >= CM * CD) return;
    const int d  = idx % CD;
    const int bn = idx / CD;
    const int n  = bn % CN;
    const float xv = x[idx];
    const float pv = (n == 0) ? 0.f : x[idx - CD];
    const size_t ro = (size_t)bn * (2 * CD);
    comb[ro + d]      = xv;
    comb[ro + CD + d] = xv - pv;
}

__global__ __launch_bounds__(256)
void rmsnorm_kernel(float* __restrict__ p, const float* __restrict__ gamma, int Dd) {
    __shared__ float red[256];
    float* pr = p + (size_t)blockIdx.x * Dd;
    float s = 0.f;
    for (int i = threadIdx.x; i < Dd; i += 256) { const float v = pr[i]; s += v * v; }
    red[threadIdx.x] = s;
    __syncthreads();
    for (int off = 128; off > 0; off >>= 1) {
        if (threadIdx.x < (unsigned)off) red[threadIdx.x] += red[threadIdx.x + off];
        __syncthreads();
    }
    const float scale = rsqrtf(red[0] / (float)Dd + CEPS);
    for (int i = threadIdx.x; i < Dd; i += 256) pr[i] = pr[i] * scale * gamma[i];
}

__global__ void conv_silu_kernel(const float* __restrict__ xz, const float* __restrict__ Wconv,
                                 const float* __restrict__ bconv, float* __restrict__ u) {
    const int idx = blockIdx.x * blockDim.x + threadIdx.x;
    if (idx >= CM * CDI) return;
    const int d  = idx % CDI;
    const int bn = idx / CDI;
    const int n  = bn % CN;
    const int b  = bn / CN;
    float acc = bconv[d];
    #pragma unroll
    for (int j = 0; j < CDC; ++j) {
        const int t = n - (CDC - 1) + j;
        if (t >= 0)
            acc += xz[((size_t)(b * CN + t)) * (2 * CDI) + d] * Wconv[d * CDC + j];
    }
    u[idx] = acc / (1.f + __expf(-acc));
}

// Selective scan: one lane per (b, d, s); 16-lane shfl_xor dot with C; fused D-skip + silu(z) gate.
__global__ __launch_bounds__(256)
void scan_kernel(const float* __restrict__ delta, const float* __restrict__ u,
                 const float* __restrict__ xdbl, const float* __restrict__ xz,
                 const float* __restrict__ A_log, const float* __restrict__ Dparam,
                 float* __restrict__ y) {
    const int tid = blockIdx.x * 256 + threadIdx.x;
    const int ch  = tid >> 4;
    const int s   = tid & 15;
    if (ch >= CB * CDI) return;
    const int b = ch / CDI;
    const int d = ch % CDI;

    const float Ads = -__expf(A_log[d * CDS + s]);
    const float Dd  = Dparam[d];
    const float* dp  = delta + (size_t)b * CN * CDI + d;
    const float* up  = u     + (size_t)b * CN * CDI + d;
    const float* bpp = xdbl  + (size_t)b * CN * (CDTR + 2 * CDS) + CDTR + s;
    const float* cpp = xdbl  + (size_t)b * CN * (CDTR + 2 * CDS) + CDTR + CDS + s;
    const float* zp  = xz    + (size_t)b * CN * (2 * CDI) + CDI + d;

    float h = 0.f;
    for (int t = 0; t < CN; ++t) {
        if (t + 8 < CN) {
            __builtin_prefetch(dp + (size_t)(t + 8) * CDI, 0, 0);
            __builtin_prefetch(up + (size_t)(t + 8) * CDI, 0, 0);
        }
        const float dt = dp[(size_t)t * CDI];
        const float ut = up[(size_t)t * CDI];
        const float bt = bpp[(size_t)t * (CDTR + 2 * CDS)];
        const float ct = cpp[(size_t)t * (CDTR + 2 * CDS)];
        h = __expf(dt * Ads) * h + (dt * ut) * bt;
        float partial = h * ct;
        partial += __shfl_xor(partial, 1, 16);
        partial += __shfl_xor(partial, 2, 16);
        partial += __shfl_xor(partial, 4, 16);
        partial += __shfl_xor(partial, 8, 16);
        if (s == 0) {
            const float zv = zp[(size_t)t * (2 * CDI)];
            const float gate = zv / (1.f + __expf(-zv));
            y[((size_t)(b * CN + t)) * CDI + d] = (partial + ut * Dd) * gate;
        }
    }
}

extern "C" void kernel_launch(void* const* d_in, const int* in_sizes, int n_in,
                              void* d_out, int out_size, void* d_ws, size_t ws_size,
                              hipStream_t stream) {
    (void)in_sizes; (void)n_in; (void)out_size; (void)ws_size;
    const float* x      = (const float*)d_in[0];
    const float* Wp     = (const float*)d_in[1];
    const float* bp     = (const float*)d_in[2];
    const float* gamma  = (const float*)d_in[3];
    const float* Win    = (const float*)d_in[4];
    const float* Wconv  = (const float*)d_in[5];
    const float* bconv  = (const float*)d_in[6];
    const float* Wx     = (const float*)d_in[7];
    const float* Wdt    = (const float*)d_in[8];
    const float* bdt    = (const float*)d_in[9];
    const float* A_log  = (const float*)d_in[10];
    const float* Dparam = (const float*)d_in[11];
    const float* Wout   = (const float*)d_in[12];
    float* out = (float*)d_out;

    float* ws    = (float*)d_ws;
    float* comb  = ws;                           // [4096,1536] ; reused as delta
    float* norm  = comb  + (size_t)CM * 2 * CD;  // [4096, 768] ; reused as x_dbl
    float* xzbuf = norm  + (size_t)CM * CD;      // [4096,3072]
    float* ubuf  = xzbuf + (size_t)CM * 2 * CDI; // [4096,1536]
    float* ybuf  = ubuf  + (size_t)CM * CDI;     // [4096,1536]
    float* deltab = comb;
    float* xdbl   = norm;
    const int XD = CDTR + 2 * CDS;               // 80

    // 1) diff + concat
    diff_concat_kernel<<<(CM * CD + 255) / 256, 256, 0, stream>>>(x, comb);

    // 2) proj = comb @ Wp^T + bp      [4096,1536]x[1536,768]  (TDM-staged)
    gemm_tdm_wmma<true, false><<<(CM / 64) * (CD / 128), 256, 0, stream>>>(
        comb, 2 * CD, Wp, 2 * CD, norm, CD, bp, nullptr, CM, CD, 2 * CD);

    // 3) RMSNorm in place
    rmsnorm_kernel<<<CM, 256, 0, stream>>>(norm, gamma, CD);

    // 4) xz = normed @ Win^T          [4096,768]x[768,3072]   (TDM-staged)
    gemm_tdm_wmma<false, false><<<(CM / 64) * (2 * CDI / 128), 256, 0, stream>>>(
        norm, CD, Win, CD, xzbuf, 2 * CDI, nullptr, nullptr, CM, 2 * CDI, CD);

    // 5) depthwise conv + SiLU -> u
    conv_silu_kernel<<<(CM * CDI + 255) / 256, 256, 0, stream>>>(xzbuf, Wconv, bconv, ubuf);

    // 6) x_dbl = u @ Wx^T             [4096,1536]x[1536,80]   (direct, Nc=80)
    {
        const int tiles = (CM / 16) * (XD / 16);
        gemm_wmma_direct<false, false><<<(tiles + 7) / 8, 256, 0, stream>>>(
            ubuf, CDI, Wx, CDI, xdbl, XD, nullptr, CM, XD, CDI);
    }

    // 7) delta = softplus(dt @ Wdt^T + bdt)   dt = x_dbl[:, :48], K=48 (direct)
    {
        const int tiles = (CM / 16) * (CDI / 16);
        gemm_wmma_direct<true, true><<<(tiles + 7) / 8, 256, 0, stream>>>(
            xdbl, XD, Wdt, CDTR, deltab, CDI, bdt, CM, CDI, CDTR);
    }

    // 8) selective scan + D-skip + silu(z) gate -> y
    scan_kernel<<<(CB * CDI * CDS) / 256, 256, 0, stream>>>(
        deltab, ubuf, xdbl, xzbuf, A_log, Dparam, ybuf);

    // 9) out = y @ Wout^T + x         [4096,1536]x[1536,768] + residual (TDM-staged)
    gemm_tdm_wmma<false, true><<<(CM / 64) * (CD / 128), 256, 0, stream>>>(
        ybuf, CDI, Wout, CDI, out, CD, nullptr, x, CM, CD, CDI);
}
